// MoE_71090298684030
// MI455X (gfx1250) — compile-verified, matrix operand order checked by reading, not run on hardware
//
#include <hip/hip_runtime.h>
#include <math.h>

#define N_TOK 8192
#define DIM   1024
#define HID   1024
#define NEXP  8
#define LMBDA 0.01f

typedef __attribute__((ext_vector_type(16))) __bf16 bf16x16;
typedef __attribute__((ext_vector_type(8)))  float  f32x8;

union Frag {
  uint4   q[2];
  bf16x16 v;
};

__device__ __forceinline__ unsigned short f2bf(float f) {
  union { float f; unsigned int u; } cv;
  cv.f = f;
  unsigned int u = cv.u;
  unsigned int r = u + 0x7FFFu + ((u >> 16) & 1u);  // round-to-nearest-even
  return (unsigned short)(r >> 16);
}

// ---------------------------------------------------------------------------
// Kernel 0: zero y, zero counters, convert x -> bf16 workspace
// ---------------------------------------------------------------------------
__global__ void prep_k(const float* __restrict__ x, float* __restrict__ y,
                       unsigned short* __restrict__ xb,
                       int* __restrict__ counts, float* __restrict__ imp) {
  if (blockIdx.x == 0 && threadIdx.x < 16) {
    counts[threadIdx.x] = 0;
    imp[threadIdx.x]    = 0.f;
  }
  size_t i = (size_t)blockIdx.x * blockDim.x + threadIdx.x;
  size_t b = i * 4;
  if (b < (size_t)N_TOK * DIM) {
    float4 xv = *(const float4*)(x + b);
    *(float4*)(y + b) = make_float4(0.f, 0.f, 0.f, 0.f);
    uint2 p;
    p.x = (unsigned)f2bf(xv.x) | ((unsigned)f2bf(xv.y) << 16);
    p.y = (unsigned)f2bf(xv.z) | ((unsigned)f2bf(xv.w) << 16);
    *(uint2*)(xb + b) = p;
  }
}

// ---------------------------------------------------------------------------
// Kernel 0b: tiled transpose + convert: W1[e,d,h] -> W1T[e,h,d] (bf16),
//            W2[e,h,d] -> W2T[e,d,h] (bf16). Both are 1024x1024 per expert.
// ---------------------------------------------------------------------------
__global__ void tconv_k(const float* __restrict__ W1, const float* __restrict__ W2,
                        unsigned short* __restrict__ w1t, unsigned short* __restrict__ w2t) {
  int bid = blockIdx.x;
  int tx  = bid & 15;  bid >>= 4;
  int ty  = bid & 15;  bid >>= 4;
  int e   = bid & 7;   bid >>= 3;
  int mat = bid;                       // 0 = W1, 1 = W2
  const float* src     = (mat ? W2 : W1) + (size_t)e * 1024 * 1024;
  unsigned short* dst  = (mat ? w2t : w1t) + (size_t)e * 1024 * 1024;
  __shared__ unsigned short tile[64][66];
  const int r0 = ty * 64, c0 = tx * 64;
  #pragma unroll
  for (int kk = 0; kk < 16; ++kk) {
    int ii = threadIdx.x + 256 * kk;
    int r = ii >> 6, c = ii & 63;
    tile[c][r] = f2bf(src[(size_t)(r0 + r) * 1024 + (c0 + c)]);   // coalesced read
  }
  __syncthreads();
  #pragma unroll
  for (int kk = 0; kk < 16; ++kk) {
    int ii = threadIdx.x + 256 * kk;
    int r = ii >> 6, c = ii & 63;
    dst[(size_t)(c0 + r) * 1024 + (r0 + c)] = tile[r][c];         // coalesced write
  }
}

// ---------------------------------------------------------------------------
// Kernel 1: gating. One wave32 per token.
// ---------------------------------------------------------------------------
__global__ void gate_k(const float* __restrict__ x, const float* __restrict__ wg_w,
                       const float* __restrict__ wg_b, float* __restrict__ gates,
                       int* __restrict__ counts, float* __restrict__ imp,
                       int* __restrict__ tok_idx) {
  const int tok  = (int)((blockIdx.x * blockDim.x + threadIdx.x) >> 5);
  const int lane = threadIdx.x & 31;
  if (tok >= N_TOK) return;
  const float* xr = x + (size_t)tok * DIM;

  float acc[NEXP];
  #pragma unroll
  for (int e = 0; e < NEXP; ++e) acc[e] = 0.f;
  for (int d = lane; d < DIM; d += 32) {
    float xv = xr[d];
    const float* wr = wg_w + d * NEXP;
    #pragma unroll
    for (int e = 0; e < NEXP; ++e) acc[e] += xv * wr[e];
  }
  #pragma unroll
  for (int e = 0; e < NEXP; ++e) {
    float a = acc[e];
    a += __shfl_xor(a, 16, 32);
    a += __shfl_xor(a, 8, 32);
    a += __shfl_xor(a, 4, 32);
    a += __shfl_xor(a, 2, 32);
    a += __shfl_xor(a, 1, 32);
    acc[e] = a + wg_b[e];
  }
  // stable top-2 (strict '>' keeps lower index on ties, matching lax.top_k)
  float b1v = -3.4e38f, b2v = -3.4e38f;
  int   i1 = 0, i2 = 0;
  #pragma unroll
  for (int e = 0; e < NEXP; ++e) {
    float l = acc[e];
    if (l > b1v) { b2v = b1v; i2 = i1; b1v = l; i1 = e; }
    else if (l > b2v) { b2v = l; i2 = e; }
  }
  float z  = __expf(b2v - b1v);
  float g0 = 1.f / (1.f + z);
  float g1 = z / (1.f + z);
  if (lane < NEXP)
    gates[(size_t)tok * NEXP + lane] = (lane == i1) ? g0 : ((lane == i2) ? g1 : 0.f);
  if (lane == 0) {
    atomicAdd(&imp[i1], g0);
    atomicAdd(&imp[i2], g1);
    int p1 = atomicAdd(&counts[i1], 1); tok_idx[i1 * N_TOK + p1] = tok;
    int p2 = atomicAdd(&counts[i2], 1); tok_idx[i2 * N_TOK + p2] = tok;
  }
}

// ---------------------------------------------------------------------------
// Kernel 2: grouped expert GEMM. Block = 32 gathered tokens of one expert,
// 128 threads (4 waves -> only 2 waves/SIMD residency in CU mode, so each
// wave gets a large VGPR budget and the pipelined WMMA loop stays spill-free).
// Per wave: 2 row-tiles x 2 col-tiles (32 accumulator VGPRs + 4 operand
// fragments). Fused  h = relu(X*W1+b1) ; y += g*(h*W2+b2).
// ---------------------------------------------------------------------------
__global__ __launch_bounds__(128, 1) void moe_gemm_k(
    const unsigned short* __restrict__ xb,
    const unsigned short* __restrict__ w1t,
    const unsigned short* __restrict__ w2t,
    const float* __restrict__ b1,
    const float* __restrict__ b2,
    const float* __restrict__ gates,
    const int* __restrict__ counts,
    const int* __restrict__ tok_idx,
    float* __restrict__ y) {
  const int e   = blockIdx.x >> 8;   // 256 tiles per expert
  const int tb  = blockIdx.x & 255;
  const int cnt = counts[e];
  const int m0  = tb * 32;
  if (m0 >= cnt) return;

  extern __shared__ char smem[];
  unsigned short* hS   = (unsigned short*)smem;                 // 32 x 1032 bf16
  int*            tokS = (int*)(smem + 32 * 1032 * 2);          // 32
  float*          gtS  = (float*)(smem + 32 * 1032 * 2 + 128);  // 32

  const int tid  = threadIdx.x;
  const int lane = tid & 31;
  const int wv   = tid >> 5;    // 0..3 : 32-column slice within the 128-col chunk
  const int hl   = lane >> 4;   // lane half (K-group select per WMMA layout)
  const int idx  = lane & 15;   // row of A-frag / col of B-frag

  if (tid < 32) {
    int slot = m0 + tid;
    int tk = 0; float g = 0.f;
    if (slot < cnt) { tk = tok_idx[e * N_TOK + slot]; g = gates[(size_t)tk * NEXP + e]; }
    tokS[tid] = tk; gtS[tid] = g;
  }
  __syncthreads();

  const unsigned short* w1e = w1t + (size_t)e * DIM * HID;
  const unsigned short* w2e = w2t + (size_t)e * HID * DIM;

  int arow[2];
  #pragma unroll
  for (int rt = 0; rt < 2; ++rt) arow[rt] = tokS[rt * 16 + idx];

  const f32x8 vzero = {0.f, 0.f, 0.f, 0.f, 0.f, 0.f, 0.f, 0.f};

  // ---------------- Phase 1: h = relu(X @ W1 + b1) ----------------
  for (int hc = 0; hc < 8; ++hc) {           // 8 chunks of 128 columns
    f32x8 acc[2][2];
    #pragma unroll
    for (int rt = 0; rt < 2; ++rt)
      #pragma unroll
      for (int ct = 0; ct < 2; ++ct) acc[rt][ct] = vzero;

    const unsigned short* ap[2];
    const unsigned short* bp[2];
    #pragma unroll
    for (int rt = 0; rt < 2; ++rt)
      ap[rt] = xb + (size_t)arow[rt] * DIM + hl * 8;
    #pragma unroll
    for (int ct = 0; ct < 2; ++ct) {
      int ncol = hc * 128 + wv * 32 + ct * 16 + idx;
      bp[ct] = w1e + (size_t)ncol * DIM + hl * 8;
    }
    for (int k = 0; k < DIM; k += 32) {
      Frag a[2], b[2];
      #pragma unroll
      for (int rt = 0; rt < 2; ++rt) {
        a[rt].q[0] = *(const uint4*)(ap[rt] + k);
        a[rt].q[1] = *(const uint4*)(ap[rt] + k + 16);
      }
      #pragma unroll
      for (int ct = 0; ct < 2; ++ct) {
        b[ct].q[0] = *(const uint4*)(bp[ct] + k);
        b[ct].q[1] = *(const uint4*)(bp[ct] + k + 16);
        __builtin_prefetch(bp[ct] + k + 128, 0, 1);
      }
      #pragma unroll
      for (int rt = 0; rt < 2; ++rt)
        #pragma unroll
        for (int ct = 0; ct < 2; ++ct)
          acc[rt][ct] = __builtin_amdgcn_wmma_f32_16x16x32_bf16(
              false, a[rt].v, false, b[ct].v, (short)0, acc[rt][ct], false, false);
    }
    #pragma unroll
    for (int rt = 0; rt < 2; ++rt)
      #pragma unroll
      for (int ct = 0; ct < 2; ++ct) {
        int col = hc * 128 + wv * 32 + ct * 16 + idx;
        float bias = b1[e * HID + col];
        #pragma unroll
        for (int r = 0; r < 8; ++r) {
          int row = rt * 16 + hl * 8 + r;   // C layout: vgpr r, lane-half selects M+8
          float v = acc[rt][ct][r] + bias;
          v = fmaxf(v, 0.f);
          hS[row * 1032 + col] = f2bf(v);
        }
      }
  }
  __syncthreads();

  // ---------------- Phase 2: out = h @ W2 + b2 ; y += g*out ----------------
  for (int dc = 0; dc < 8; ++dc) {           // 8 chunks of 128 columns
    f32x8 acc[2][2];
    #pragma unroll
    for (int rt = 0; rt < 2; ++rt)
      #pragma unroll
      for (int ct = 0; ct < 2; ++ct) acc[rt][ct] = vzero;

    const unsigned short* ap[2];
    const unsigned short* bp[2];
    #pragma unroll
    for (int rt = 0; rt < 2; ++rt)
      ap[rt] = hS + (rt * 16 + idx) * 1032 + hl * 8;
    #pragma unroll
    for (int ct = 0; ct < 2; ++ct) {
      int dcol = dc * 128 + wv * 32 + ct * 16 + idx;
      bp[ct] = w2e + (size_t)dcol * HID + hl * 8;
    }
    for (int k = 0; k < HID; k += 32) {
      Frag a[2], b[2];
      #pragma unroll
      for (int rt = 0; rt < 2; ++rt) {
        a[rt].q[0] = *(const uint4*)(ap[rt] + k);      // ds_load_b128 from LDS h
        a[rt].q[1] = *(const uint4*)(ap[rt] + k + 16);
      }
      #pragma unroll
      for (int ct = 0; ct < 2; ++ct) {
        b[ct].q[0] = *(const uint4*)(bp[ct] + k);
        b[ct].q[1] = *(const uint4*)(bp[ct] + k + 16);
        __builtin_prefetch(bp[ct] + k + 128, 0, 1);
      }
      #pragma unroll
      for (int rt = 0; rt < 2; ++rt)
        #pragma unroll
        for (int ct = 0; ct < 2; ++ct)
          acc[rt][ct] = __builtin_amdgcn_wmma_f32_16x16x32_bf16(
              false, a[rt].v, false, b[ct].v, (short)0, acc[rt][ct], false, false);
    }
    #pragma unroll
    for (int rt = 0; rt < 2; ++rt)
      #pragma unroll
      for (int ct = 0; ct < 2; ++ct) {
        int dcol = dc * 128 + wv * 32 + ct * 16 + idx;
        float bias = b2[e * DIM + dcol];
        #pragma unroll
        for (int r = 0; r < 8; ++r) {
          int row = rt * 16 + hl * 8 + r;
          if (m0 + row < cnt) {
            float v = acc[rt][ct][r] + bias;
            atomicAdd(&y[(size_t)tokS[row] * DIM + dcol], gtS[row] * v);
          }
        }
      }
  }
}

// ---------------------------------------------------------------------------
// Kernel 3: load-balancing loss = lmbda * std(importance, ddof=1)/mean
// ---------------------------------------------------------------------------
__global__ void loss_k(const float* __restrict__ imp, float* __restrict__ loss) {
  if (blockIdx.x == 0 && threadIdx.x == 0) {
    float v[NEXP];
    float mean = 0.f;
    #pragma unroll
    for (int e = 0; e < NEXP; ++e) { v[e] = imp[e] / (float)N_TOK; mean += v[e]; }
    mean *= (1.f / (float)NEXP);
    float var = 0.f;
    #pragma unroll
    for (int e = 0; e < NEXP; ++e) { float d = v[e] - mean; var += d * d; }
    var *= (1.f / (float)(NEXP - 1));
    loss[0] = LMBDA * sqrtf(var) / mean;
  }
}

// ---------------------------------------------------------------------------
extern "C" void kernel_launch(void* const* d_in, const int* in_sizes, int n_in,
                              void* d_out, int out_size, void* d_ws, size_t ws_size,
                              hipStream_t stream) {
  (void)in_sizes; (void)n_in; (void)out_size; (void)ws_size;
  const float* x    = (const float*)d_in[0];
  const float* wg_w = (const float*)d_in[1];
  const float* wg_b = (const float*)d_in[2];
  const float* W1   = (const float*)d_in[3];
  const float* b1   = (const float*)d_in[4];
  const float* W2   = (const float*)d_in[5];
  const float* b2   = (const float*)d_in[6];

  float* y     = (float*)d_out;                       // [N, D]
  float* loss  = y + (size_t)N_TOK * DIM;             // [1]
  float* gates = loss + 1;                            // [N, E]

  char* ws = (char*)d_ws;
  int*            counts = (int*)ws;                                    // 16 ints
  float*          imp    = (float*)(ws + 64);                           // 16 floats
  int*            tok    = (int*)(ws + 1024);                           // E*N ints (256 KB)
  unsigned short* xb     = (unsigned short*)(ws + 1024 + (size_t)NEXP * N_TOK * 4);
  unsigned short* w1t    = xb  + (size_t)N_TOK * DIM;                   // 16 MB
  unsigned short* w2t    = w1t + (size_t)NEXP * DIM * HID;              // 16 MB

  const int SMEM_BYTES = 32 * 1032 * 2 + 128 + 128;   // 66,304 B (< 320 KB WGP LDS)
  hipFuncSetAttribute(reinterpret_cast<const void*>(&moe_gemm_k),
                      hipFuncAttributeMaxDynamicSharedMemorySize, SMEM_BYTES);

  prep_k <<<(N_TOK * DIM) / (4 * 256), 256, 0, stream>>>(x, y, xb, counts, imp);
  tconv_k<<<2 * NEXP * 16 * 16,        256, 0, stream>>>(W1, W2, w1t, w2t);
  gate_k <<<(N_TOK * 32) / 256,        256, 0, stream>>>(x, wg_w, wg_b, gates, counts, imp, tok);
  moe_gemm_k<<<NEXP * 256, 128, SMEM_BYTES, stream>>>(xb, w1t, w2t, b1, b2, gates,
                                                      counts, tok, y);
  loss_k <<<1, 32, 0, stream>>>(imp, loss);
}
